// LeNet_14096082665603
// MI455X (gfx1250) — matrix operand order repaired and checked by reading, not compiled, not run on hardware
//
#include <hip/hip_runtime.h>
#include <cstdint>
#include <cstddef>

typedef __attribute__((ext_vector_type(16))) _Float16 v16h;
typedef __attribute__((ext_vector_type(8)))  _Float16 v8h;
typedef __attribute__((ext_vector_type(8)))  float    v8f;

#define B_TOTAL 8192

// sigmoid = rcp(1 + exp(-x)) : v_exp_f32 + v_add_f32 + v_rcp_f32 (no IEEE div chain)
__device__ __forceinline__ float sigmoidf_(float x) {
    return __builtin_amdgcn_rcpf(1.0f + __expf(-x));
}

// ---------------------------------------------------------------------------
// Weight prep: W is [N][K] row-major f32. Emit f16 pre-swizzled into the
// per-lane B-fragment order for v_wmma_f32_16x16x32_f16:
//   dst[((nt*Kchunks + ch)*32 + lane)*16 + j] = W[nt*16 + (lane&15)]
//                                                [ch*32 + (lane<16?0:8) + j + (j>=8?8:0)]
// zero-padded outside [N,K].
// ---------------------------------------------------------------------------
__global__ void prep_w_kernel(const float* __restrict__ W, _Float16* __restrict__ dst,
                              int N, int K, int Kchunks, int total) {
    int idx = blockIdx.x * blockDim.x + threadIdx.x;
    if (idx >= total) return;
    int j    = idx & 15;
    int lane = (idx >> 4) & 31;
    int t    = idx >> 9;            // nt*Kchunks + ch
    int ch   = t % Kchunks;
    int nt   = t / Kchunks;
    int n = nt * 16 + (lane & 15);
    int k = ch * 32 + ((lane < 16) ? 0 : 8) + j + ((j >= 8) ? 8 : 0);
    float v = (n < N && k < K) ? W[(size_t)n * K + k] : 0.0f;
    dst[idx] = (_Float16)v;
}

// ---------------------------------------------------------------------------
// conv1: x[B,1,28,28] f32, K1 swizzled (N=6->16, K=25->32). Implicit im2col,
// WMMA 16x16x32 f16, fused sigmoid -> LDS -> 2x2 avgpool -> P1h[B,6,14,14] f16.
// One block (8 waves) per image; 49 M-tiles.
// ---------------------------------------------------------------------------
__global__ void conv1_kernel(const float* __restrict__ x,
                             const _Float16* __restrict__ B1f,
                             _Float16* __restrict__ P1h) {
    __shared__ _Float16 sH[784 * 16];
    const int b    = blockIdx.x;
    const int tid  = threadIdx.x;
    const int wave = tid >> 5;
    const int lane = tid & 31;
    const int m     = lane & 15;
    const int kbase = (lane < 16) ? 0 : 8;   // A-fragment K base (doc layout)
    const int mAdd  = (lane >> 4) << 3;      // C-fragment M offset (0 or 8)
    const float* img = x + (size_t)b * 784;

    v16h bfrag = *(const v16h*)(B1f + lane * 16);

    for (int tile = wave; tile < 49; tile += 8) {
        int p  = tile * 16 + m;
        int oy = p / 28, ox = p % 28;
        v16h a;
        #pragma unroll
        for (int j = 0; j < 16; ++j) {
            int k = kbase + j + ((j >= 8) ? 8 : 0);   // j<8: kbase+j ; j>=8: kbase+16+(j-8)
            float v = 0.0f;
            if (k < 25) {
                int kh = k / 5, kw = k % 5;
                int iy = oy + kh - 2, ix = ox + kw - 2;
                if (iy >= 0 && iy < 28 && ix >= 0 && ix < 28)
                    v = img[iy * 28 + ix];
            }
            a[j] = (_Float16)v;
        }
        v8f c = {};
        c = __builtin_amdgcn_wmma_f32_16x16x32_f16(false, a, false, bfrag,
                                                   (short)0, c, false, false);
        #pragma unroll
        for (int r = 0; r < 8; ++r) {
            int pix = tile * 16 + r + mAdd;
            sH[pix * 16 + (lane & 15)] = (_Float16)sigmoidf_(c[r]);
        }
    }
    __syncthreads();
    // 2x2 average pool -> [6][14][14]
    for (int idx = tid; idx < 1176; idx += 256) {
        int cch = idx / 196;
        int rem = idx % 196;
        int y = rem / 14, xo = rem % 14;
        int p00 = (2 * y) * 28 + 2 * xo;
        float s = (float)sH[p00 * 16 + cch] + (float)sH[(p00 + 1) * 16 + cch]
                + (float)sH[(p00 + 28) * 16 + cch] + (float)sH[(p00 + 29) * 16 + cch];
        P1h[(size_t)b * 1176 + idx] = (_Float16)(0.25f * s);
    }
}

// ---------------------------------------------------------------------------
// conv2: P1h[B,6,14,14] f16 -> conv 5x5 (no pad) N=16, K=150->160 (5 chunks),
// fused sigmoid -> LDS -> 2x2 avgpool -> P2h[B][416] f16 (400 valid + zeros).
// One block per image; waves 0..6 cover the 7 M-tiles (100 pixels).
// ---------------------------------------------------------------------------
__global__ void conv2_kernel(const _Float16* __restrict__ P1h,
                             const _Float16* __restrict__ B2f,
                             _Float16* __restrict__ P2h) {
    __shared__ _Float16 sH[112 * 16];
    const int b    = blockIdx.x;
    const int tid  = threadIdx.x;
    const int wave = tid >> 5;
    const int lane = tid & 31;
    const _Float16* in = P1h + (size_t)b * 1176;

    if (wave < 7) {
        const int tile  = wave;
        const int m     = lane & 15;
        const int kbase = (lane < 16) ? 0 : 8;
        const int mAdd  = (lane >> 4) << 3;
        int p = tile * 16 + m;
        bool valid = (p < 100);
        int oy = p / 10, ox = p % 10;
        v8f c = {};
        #pragma unroll
        for (int ch = 0; ch < 5; ++ch) {
            v16h a;
            #pragma unroll
            for (int j = 0; j < 16; ++j) {
                int k = ch * 32 + kbase + j + ((j >= 8) ? 8 : 0);
                _Float16 v = (_Float16)0.0f;
                if (valid && k < 150) {
                    int ci = k / 25, r2 = k % 25;
                    int kh = r2 / 5, kw = r2 % 5;
                    v = in[ci * 196 + (oy + kh) * 14 + (ox + kw)];
                }
                a[j] = v;
            }
            v16h bfrag = *(const v16h*)(B2f + (size_t)(ch * 32 + lane) * 16);
            c = __builtin_amdgcn_wmma_f32_16x16x32_f16(false, a, false, bfrag,
                                                       (short)0, c, false, false);
        }
        #pragma unroll
        for (int r = 0; r < 8; ++r) {
            int pix = tile * 16 + r + mAdd;
            sH[pix * 16 + (lane & 15)] = (_Float16)sigmoidf_(c[r]);
        }
    }
    __syncthreads();
    for (int idx = tid; idx < 416; idx += 256) {
        _Float16 out = (_Float16)0.0f;
        if (idx < 400) {
            int cch = idx / 25;
            int rem = idx % 25;
            int y = rem / 5, xo = rem % 5;
            int p00 = (2 * y) * 10 + 2 * xo;
            float s = (float)sH[p00 * 16 + cch] + (float)sH[(p00 + 1) * 16 + cch]
                    + (float)sH[(p00 + 10) * 16 + cch] + (float)sH[(p00 + 11) * 16 + cch];
            out = (_Float16)(0.25f * s);
        }
        P2h[(size_t)b * 416 + idx] = out;
    }
}

// ---------------------------------------------------------------------------
// Generic FC layer: out[M][Npad] = act(A[M][Kpad] * W^T + bias)
// A row-major f16 (Kpad = KCHUNKS*32), Bf pre-swizzled fragments.
// Block = one 16-row batch tile; wave w = n-tile w (blockDim = Ntiles*32).
// KCHUNKS is compile-time so the WMMA chain fully unrolls and fragment loads
// software-pipeline against the matrix ops.
// ---------------------------------------------------------------------------
template <int KCHUNKS, bool SIGMOID>
__global__ void fc_kernel(const _Float16* __restrict__ A,
                          const _Float16* __restrict__ Bf,
                          const float* __restrict__ bias,
                          _Float16* __restrict__ outH,
                          float* __restrict__ outF,
                          int Nvalid, int outStride) {
    const int row0 = blockIdx.x * 16;
    const int wave = threadIdx.x >> 5;      // n-tile index
    const int lane = threadIdx.x & 31;
    const int m    = lane & 15;
    const int aoff = (lane < 16) ? 0 : 8;
    const int mAdd = (lane >> 4) << 3;
    const int row  = row0 + m;
    const _Float16* arow = A + (size_t)row * (KCHUNKS * 32) + aoff;
    const _Float16* bfp  = Bf + ((size_t)wave * KCHUNKS * 32 + lane) * 16;

    v8f c = {};
    #pragma unroll
    for (int ch = 0; ch < KCHUNKS; ++ch) {
        union { v16h v; v8h h[2]; } u;
        u.h[0] = *(const v8h*)(arow + ch * 32);        // K = kbase + 0..7
        u.h[1] = *(const v8h*)(arow + ch * 32 + 16);   // K = kbase + 16..23
        v16h bfrag = *(const v16h*)(bfp + (size_t)ch * 512);
        c = __builtin_amdgcn_wmma_f32_16x16x32_f16(false, u.v, false, bfrag,
                                                   (short)0, c, false, false);
    }
    const int n  = wave * 16 + (lane & 15);
    const float bn = (n < Nvalid) ? bias[n] : 0.0f;
    #pragma unroll
    for (int r = 0; r < 8; ++r) {
        int orow = row0 + r + mAdd;
        float v = c[r] + bn;
        if (SIGMOID) {
            outH[(size_t)orow * outStride + n] = (_Float16)sigmoidf_(v);
        } else {
            outF[(size_t)orow * outStride + n] = v;
        }
    }
}

// ---------------------------------------------------------------------------
// Row softmax over 10 logits (stored with stride 16 in f32 scratch).
// ---------------------------------------------------------------------------
__global__ void softmax_kernel(const float* __restrict__ logits,
                               float* __restrict__ out) {
    int i = blockIdx.x * blockDim.x + threadIdx.x;
    if (i >= B_TOTAL) return;
    const float* l = logits + (size_t)i * 16;
    float mx = l[0];
    #pragma unroll
    for (int j = 1; j < 10; ++j) mx = fmaxf(mx, l[j]);
    float e[10];
    float s = 0.0f;
    #pragma unroll
    for (int j = 0; j < 10; ++j) { e[j] = __expf(l[j] - mx); s += e[j]; }
    float inv = __builtin_amdgcn_rcpf(s);
    #pragma unroll
    for (int j = 0; j < 10; ++j) out[(size_t)i * 10 + j] = e[j] * inv;
}

// ---------------------------------------------------------------------------
extern "C" void kernel_launch(void* const* d_in, const int* in_sizes, int n_in,
                              void* d_out, int out_size, void* d_ws, size_t ws_size,
                              hipStream_t stream) {
    (void)in_sizes; (void)n_in; (void)out_size; (void)ws_size;
    const float* x  = (const float*)d_in[0];
    const float* K1 = (const float*)d_in[1];
    const float* K2 = (const float*)d_in[2];
    const float* W1 = (const float*)d_in[3];
    const float* b1 = (const float*)d_in[4];
    const float* W2 = (const float*)d_in[5];
    const float* b2 = (const float*)d_in[6];
    const float* W3 = (const float*)d_in[7];
    const float* b3 = (const float*)d_in[8];

    char* ws = (char*)d_ws;
    size_t cur = 0;
    auto alloc = [&](size_t bytes) -> void* {
        void* p = ws + cur;
        cur = (cur + bytes + 255) & ~(size_t)255;
        return p;
    };
    _Float16* B1f = (_Float16*)alloc((size_t)1 * 1 * 512 * 2);   // conv1 weights
    _Float16* B2f = (_Float16*)alloc((size_t)1 * 5 * 512 * 2);   // conv2 weights
    _Float16* W1f = (_Float16*)alloc((size_t)8 * 13 * 512 * 2);  // fc1 weights
    _Float16* W2f = (_Float16*)alloc((size_t)6 * 4 * 512 * 2);   // fc2 weights
    _Float16* W3f = (_Float16*)alloc((size_t)1 * 3 * 512 * 2);   // fc3 weights
    _Float16* P1h = (_Float16*)alloc((size_t)B_TOTAL * 1176 * 2); // [B,6,14,14]
    _Float16* P2h = (_Float16*)alloc((size_t)B_TOTAL * 416 * 2);  // [B,416]
    _Float16* A2  = (_Float16*)alloc((size_t)B_TOTAL * 128 * 2);  // [B,128]
    _Float16* A3  = (_Float16*)alloc((size_t)B_TOTAL * 96 * 2);   // [B,96]
    float*    LG  = (float*)alloc((size_t)B_TOTAL * 16 * 4);      // [B,16] logits

    auto launch_prep = [&](const float* W, _Float16* dst, int N, int K,
                           int Kchunks, int Ntiles) {
        int total = Ntiles * Kchunks * 512;
        prep_w_kernel<<<(total + 255) / 256, 256, 0, stream>>>(W, dst, N, K, Kchunks, total);
    };
    launch_prep(K1, B1f,   6,  25,  1, 1);
    launch_prep(K2, B2f,  16, 150,  5, 1);
    launch_prep(W1, W1f, 120, 400, 13, 8);
    launch_prep(W2, W2f,  84, 120,  4, 6);
    launch_prep(W3, W3f,  10,  84,  3, 1);

    conv1_kernel<<<B_TOTAL, 256, 0, stream>>>(x, B1f, P1h);
    conv2_kernel<<<B_TOTAL, 256, 0, stream>>>(P1h, B2f, P2h);
    fc_kernel<13, true ><<<B_TOTAL / 16, 256, 0, stream>>>(P2h, W1f, b1, A2, nullptr, 120, 128);
    fc_kernel< 4, true ><<<B_TOTAL / 16, 192, 0, stream>>>(A2,  W2f, b2, A3, nullptr,  84,  96);
    fc_kernel< 3, false><<<B_TOTAL / 16,  32, 0, stream>>>(A3,  W3f, b3, nullptr, LG,  10,  16);
    softmax_kernel<<<(B_TOTAL + 255) / 256, 256, 0, stream>>>(LG, (float*)d_out);
}
